// SEAM_59141699666564
// MI455X (gfx1250) — matrix operand order repaired and checked by reading, not compiled, not run on hardware
//
#include <hip/hip_runtime.h>
#include <hip/hip_bf16.h>

#define DIMC  768
#define HNUM  12
#define SNUM  16
#define DHEAD 64
#define NSEQ  4096
#define BNUM  8
#define K2    192   // H*S
static constexpr float EPS = 1e-8f;

typedef __attribute__((ext_vector_type(16))) __bf16 v16bf;
typedef __attribute__((ext_vector_type(8)))  float  v8f;

// ---------------------------------------------------------------------------
// Pass 0: mk1[d][s] = sum_e h1[d][e]*mk[s][e]   (64x16)
//         mvh2[s][e] = sum_d mv[s][d]*h2[d][e]  (16x64)
// ---------------------------------------------------------------------------
__global__ __launch_bounds__(256) void k_prep(
    const float* __restrict__ h1, const float* __restrict__ mk,
    const float* __restrict__ mv, const float* __restrict__ h2,
    float* __restrict__ mk1, float* __restrict__ mvh2) {
  int t = threadIdx.x;
  for (int i = t; i < 1024; i += 256) {
    int d = i >> 4, s = i & 15;
    float a = 0.f;
    for (int e = 0; e < 64; ++e) a += h1[d * 64 + e] * mk[s * 64 + e];
    mk1[i] = a;                       // [d*16+s]
  }
  for (int i = t; i < 1024; i += 256) {
    int s = i >> 6, e = i & 63;
    float a = 0.f;
    for (int d = 0; d < 64; ++d) a += mv[s * 64 + d] * h2[d * 64 + e];
    mvh2[i] = a;                      // [s*64+e]
  }
}

// ---------------------------------------------------------------------------
// Pass 1: logits = x_head @ mk1, softmax over S, store p (bf16) in
// [b][n][h*16+s] layout; accumulate colsum[b][h][s] over N via shuffle+atomic.
// Block: 384 threads = 12 waves (one head per wave, lane = row). 32 rows/block.
// ---------------------------------------------------------------------------
__global__ __launch_bounds__(384) void k_attn(
    const float* __restrict__ x, const float* __restrict__ mk1g,
    __hip_bfloat16* __restrict__ p, float* __restrict__ colsum) {
  extern __shared__ float sm1[];
  float* xs   = sm1;                 // 32 rows * 769 (odd stride: bank-safe)
  float* mk1s = sm1 + 32 * 769;      // 1024
  const int t  = threadIdx.x;
  const int b  = blockIdx.y;
  const int n0 = blockIdx.x * 32;

  const float* xsrc = x + ((size_t)b * NSEQ + n0) * DIMC;
  for (int i = t; i < 6144; i += 384) {          // 32*768 floats as float4
    float4 g = ((const float4*)xsrc)[i];
    int r = (i * 4) / DIMC, c = (i * 4) % DIMC;
    float* d = &xs[r * 769 + c];
    d[0] = g.x; d[1] = g.y; d[2] = g.z; d[3] = g.w;
  }
  for (int i = t; i < 1024; i += 384) mk1s[i] = mk1g[i];
  __syncthreads();

  const int h = t >> 5;              // head (wave)
  const int r = t & 31;              // row (lane)
  float acc[SNUM];
#pragma unroll
  for (int s = 0; s < SNUM; ++s) acc[s] = 0.f;
  const float* xr = &xs[r * 769 + h * DHEAD];
#pragma unroll 8
  for (int d = 0; d < DHEAD; ++d) {
    float xv = xr[d];
#pragma unroll
    for (int s = 0; s < SNUM; ++s) acc[s] += xv * mk1s[d * SNUM + s];
  }
  // softmax over S
  float mx = acc[0];
#pragma unroll
  for (int s = 1; s < SNUM; ++s) mx = fmaxf(mx, acc[s]);
  float sum = 0.f;
#pragma unroll
  for (int s = 0; s < SNUM; ++s) { acc[s] = __expf(acc[s] - mx); sum += acc[s]; }
  float inv = 1.f / sum;
#pragma unroll
  for (int s = 0; s < SNUM; ++s) acc[s] *= inv;

  size_t base = ((size_t)(b * NSEQ + n0 + r)) * K2 + h * SNUM;
#pragma unroll
  for (int s = 0; s < SNUM; ++s) p[base + s] = __float2bfloat16(acc[s]);

  // column sums over the 32 rows of this wave, then one atomic per (h,s)
#pragma unroll
  for (int s = 0; s < SNUM; ++s) {
    float v = acc[s];
    for (int off = 16; off > 0; off >>= 1) v += __shfl_xor(v, off);
    if (r == 0) atomicAdd(&colsum[(b * HNUM + h) * SNUM + s], v);
  }
}

// ---------------------------------------------------------------------------
// Pass 2: W3T[b][c][k=h*16+s] = (1/(colsum+eps)) * sum_d mvh2[s][d]*proj_w[c][h*64+d]
// stored bf16, k-contiguous (GEMM B-operand layout). Block = 192 thr (one k each).
// ---------------------------------------------------------------------------
__global__ __launch_bounds__(192) void k_w3(
    const float* __restrict__ proj_w, const float* __restrict__ mvh2,
    const float* __restrict__ colsum, __hip_bfloat16* __restrict__ w3t) {
  __shared__ float pw[DIMC];
  __shared__ float mh[1024];
  const int c = blockIdx.x, b = blockIdx.y, t = threadIdx.x;
  for (int i = t; i < DIMC; i += 192) pw[i] = proj_w[(size_t)c * DIMC + i];
  for (int i = t; i < 1024; i += 192) mh[i] = mvh2[i];
  float cs = colsum[b * K2 + t];
  __syncthreads();
  const int h = t >> 4, s = t & 15;
  const float* mr = &mh[s * DHEAD];
  const float* pr = &pw[h * DHEAD];
  float a = 0.f;
#pragma unroll 8
  for (int d = 0; d < DHEAD; ++d) a += mr[d] * pr[d];
  a /= (cs + EPS);
  w3t[((size_t)b * DIMC + c) * K2 + t] = __float2bfloat16(a);
}

// ---------------------------------------------------------------------------
// Pass 3: out[b] = P[b](4096x192) @ W3[b](192x768) + proj_b + x   (bf16 WMMA)
// Block tile 128x64, entire K=192 staged in LDS, 8 waves x (2x2) 16x16 tiles.
// ---------------------------------------------------------------------------
#define SA 200   // LDS row stride (bf16) for A and B tiles; 16B-aligned rows

__device__ __forceinline__ v16bf load_a(const unsigned short* As, int m, int k0, int hf) {
  // 16-bit A 16x32 layout: lanes<16 -> dwords {0..3, 8..11}; lanes>=16 -> {4..7, 12..15}
  const unsigned* bp = (const unsigned*)(As + m * SA + k0);
  union { uint4 u[2]; v16bf v; } u;
  u.u[0] = *(const uint4*)(bp + hf * 4);
  u.u[1] = *(const uint4*)(bp + hf * 4 + 8);
  return u.v;
}
__device__ __forceinline__ v16bf load_b(const unsigned short* Bs, int n, int k0, int hf) {
  // B 32x16: lane gives N=lane%16; same K order as A: lanes<16 -> dwords {0..3, 8..11}; lanes>=16 -> {4..7, 12..15}
  const unsigned* bp = (const unsigned*)(Bs + n * SA + k0);
  union { uint4 u[2]; v16bf v; } u;
  u.u[0] = *(const uint4*)(bp + hf * 4);
  u.u[1] = *(const uint4*)(bp + hf * 4 + 8);
  return u.v;
}
__device__ __forceinline__ v8f wmma_bf16(v16bf a, v16bf b, v8f c) {
  return __builtin_amdgcn_wmma_f32_16x16x32_bf16(false, a, false, b, (short)0, c, false, false);
}

__global__ __launch_bounds__(256) void k_gemm(
    const __hip_bfloat16* __restrict__ p, const __hip_bfloat16* __restrict__ w3t,
    const float* __restrict__ proj_b, const float* __restrict__ x,
    float* __restrict__ out) {
  extern __shared__ char smem[];
  unsigned short* As = (unsigned short*)smem;     // 128 x SA bf16
  unsigned short* Bs = As + 128 * SA;             // 64  x SA bf16
  float*          Cs = (float*)smem;              // overlay: 128 x 68 f32

  const int t  = threadIdx.x;
  const int cb = blockIdx.x * 64;
  const int rb = blockIdx.y * 128;
  const int b  = blockIdx.z;

  // A tile: p rows rb..rb+127, 192 bf16 each (contiguous block) -> LDS, padded rows
  const uint4* ag = (const uint4*)(p + ((size_t)(b * NSEQ + rb)) * K2);
#pragma unroll
  for (int i = 0; i < 12; ++i) {
    int u = t + i * 256;                 // 3072 uint4 total, 24 per row
    uint4 v = ag[u];
    int r = u / 24, c = u % 24;
    *(uint4*)(As + r * SA + c * 8) = v;
  }
  // B tile: W3T rows cb..cb+63 (n-major, k-contiguous)
  const uint4* bg = (const uint4*)(w3t + ((size_t)(b * DIMC + cb)) * K2);
#pragma unroll
  for (int i = 0; i < 6; ++i) {
    int u = t + i * 256;                 // 1536 uint4
    uint4 v = bg[u];
    int r = u / 24, c = u % 24;
    *(uint4*)(Bs + r * SA + c * 8) = v;
  }
  __syncthreads();

  const int lane = t & 31, w = t >> 5;
  const int wr = w >> 1, wc = w & 1;     // wave grid 4x2
  const int mW = wr * 32, nW = wc * 32;
  const int hf = lane >> 4, l16 = lane & 15;

  v8f a00{}, a01{}, a10{}, a11{};
#pragma unroll
  for (int kt = 0; kt < 6; ++kt) {
    int k0 = kt * 32;
    v16bf fa0 = load_a(As, mW + l16,      k0, hf);
    v16bf fa1 = load_a(As, mW + 16 + l16, k0, hf);
    v16bf fb0 = load_b(Bs, nW + l16,      k0, hf);
    v16bf fb1 = load_b(Bs, nW + 16 + l16, k0, hf);
    a00 = wmma_bf16(fa0, fb0, a00);
    a01 = wmma_bf16(fa0, fb1, a01);
    a10 = wmma_bf16(fa1, fb0, a10);
    a11 = wmma_bf16(fa1, fb1, a11);
  }
  __syncthreads();                       // done reading As before C overlay

  // C layout: lanes<16: M=r, N=lane ; lanes>=16: M=8+r, N=lane-16
#pragma unroll
  for (int r = 0; r < 8; ++r) {
    int mo = r + 8 * hf;
    Cs[(mW + mo) * 68      + nW + l16]      = a00[r];
    Cs[(mW + mo) * 68      + nW + 16 + l16] = a01[r];
    Cs[(mW + 16 + mo) * 68 + nW + l16]      = a10[r];
    Cs[(mW + 16 + mo) * 68 + nW + 16 + l16] = a11[r];
  }
  __syncthreads();

  // epilogue: coalesced float4 stores with bias + residual
  const float* xg = x   + ((size_t)(b * NSEQ + rb)) * DIMC + cb;
  float*       og = out + ((size_t)(b * NSEQ + rb)) * DIMC + cb;
#pragma unroll
  for (int i = 0; i < 8; ++i) {
    int u = t + i * 256;                 // 2048 float4 = 128 x 64 floats
    int r = u >> 4, cw = (u & 15) * 4;
    float4 xr = *(const float4*)(xg + (size_t)r * DIMC + cw);
    float4 pb = *(const float4*)(proj_b + cb + cw);
    const float* cr = &Cs[r * 68 + cw];
    float4 o;
    o.x = cr[0] + pb.x + xr.x;
    o.y = cr[1] + pb.y + xr.y;
    o.z = cr[2] + pb.z + xr.z;
    o.w = cr[3] + pb.w + xr.w;
    *(float4*)(og + (size_t)r * DIMC + cw) = o;
  }
}

// ---------------------------------------------------------------------------
extern "C" void kernel_launch(void* const* d_in, const int* in_sizes, int n_in,
                              void* d_out, int out_size, void* d_ws, size_t ws_size,
                              hipStream_t stream) {
  (void)in_sizes; (void)n_in; (void)out_size; (void)ws_size;
  const float* x      = (const float*)d_in[0];
  const float* h1     = (const float*)d_in[1];
  const float* mk     = (const float*)d_in[2];
  const float* mv     = (const float*)d_in[3];
  const float* h2     = (const float*)d_in[4];
  const float* proj_w = (const float*)d_in[5];
  const float* proj_b = (const float*)d_in[6];
  float* out = (float*)d_out;

  char* ws = (char*)d_ws;
  float*          colsum = (float*)ws;                         // 1536 f32
  float*          mk1    = (float*)(ws + 8192);                // 1024 f32
  float*          mvh2   = (float*)(ws + 12288);               // 1024 f32
  __hip_bfloat16* w3t    = (__hip_bfloat16*)(ws + 16384);      // 8*768*192 bf16
  __hip_bfloat16* pbuf   = (__hip_bfloat16*)(ws + 16384 + (size_t)BNUM * DIMC * K2 * 2);

  hipMemsetAsync(colsum, 0, BNUM * HNUM * SNUM * sizeof(float), stream);
  k_prep<<<1, 256, 0, stream>>>(h1, mk, mv, h2, mk1, mvh2);
  k_attn<<<dim3(NSEQ / 32, BNUM), 384, (32 * 769 + 1024) * sizeof(float), stream>>>(
      x, mk1, pbuf, colsum);
  k_w3<<<dim3(DIMC, BNUM), 192, 0, stream>>>(proj_w, mvh2, colsum, w3t);
  k_gemm<<<dim3(DIMC / 64, NSEQ / 128, BNUM), 256, (128 + 64) * SA * 2, stream>>>(
      pbuf, w3t, proj_b, x, out);
}
